// CrossModalFusionCore_87308095193476
// MI455X (gfx1250) — matrix operand order repaired and checked by reading, not compiled, not run on hardware
//
#include <hip/hip_runtime.h>
#include <hip/hip_bf16.h>

// CrossModalFusionCore for MI455X (gfx1250, wave32).
//
// Structural collapse of the reference:
//  * seq_att (K=V=broadcast graph vec): softmax uniform -> per-batch vector
//  * graph_att (Q=broadcast graph vec): identical attention for all S queries
// so the [B,S,D] output is a per-batch D-vector broadcast across S.
//
// Streams: seq read twice (2x32MB, second pass L2-hot) by two WMMA kernels with
// fragments loaded DIRECTLY from global (no LDS staging: no cross-wave reuse
// exists, and half-wave fragment runs tile full 128B cachelines); weights read
// once (32MB) by batch-amortized row-parallel GEMVs; 32MB non-temporal
// broadcast store. ~130MB @ 23.3 TB/s => ~6 us.

typedef __attribute__((ext_vector_type(16))) _Float16 v16h;
typedef __attribute__((ext_vector_type(8)))  _Float16 v8h;
typedef __attribute__((ext_vector_type(8)))  float    v8f;
typedef __attribute__((ext_vector_type(4)))  float    v4f;   // native vec for nontemporal
typedef _Float16 half_t;

constexpr int Bn = 8, Sn = 1024, Dn = 1024, Hn = 16, HDn = 64;
constexpr int VP = 1028;   // LDS pitch (f32) for [8][1024] batch vectors (1028%64=4)
constexpr float EPS = 1e-5f;

// ---------------------------------------------------------------- reductions
__device__ __forceinline__ float block_reduce_sum(float v, float* red) {
    int t = threadIdx.x;
    red[t] = v; __syncthreads();
    for (int s = 128; s > 0; s >>= 1) {
        if (t < s) red[t] += red[t + s];
        __syncthreads();
    }
    float r = red[0]; __syncthreads();
    return r;
}
__device__ __forceinline__ float block_reduce_max(float v, float* red) {
    int t = threadIdx.x;
    red[t] = v; __syncthreads();
    for (int s = 128; s > 0; s >>= 1) {
        if (t < s) red[t] = fmaxf(red[t], red[t + s]);
        __syncthreads();
    }
    float r = red[0]; __syncthreads();
    return r;
}

// ---------------------------------------------------------------- k_qgvg
// qg[b,o] = Wq[o,:].g_b + bq[o] ; vg[b,o] = Wv[o,:].g_b + bv[o]
// Grid 32 x 256: block owns 32 rows; thread = (row, batch); weights read once.
__global__ void k_qgvg(const float* __restrict__ graph,
                       const float* __restrict__ ipw, const float* __restrict__ ipb,
                       float* __restrict__ qg, float* __restrict__ vg)
{
    __shared__ float s_g[Bn][VP];
    const int t = threadIdx.x;
    for (int r = 0; r < 32; ++r) {
        const int idx = r * 256 + t;                 // 0..8191
        s_g[idx >> 10][idx & 1023] = graph[idx];
    }
    __syncthreads();

    const int o = blockIdx.x * 32 + (t >> 3);
    const int b = t & 7;
    const float4* wq = (const float4*)(ipw + (size_t)o * Dn);
    const float4* wv = (const float4*)(ipw + (size_t)(2 * Dn + o) * Dn);
    float aq = ipb[o], av = ipb[2 * Dn + o];
    for (int d4 = 0; d4 < Dn / 4; ++d4) {
        const float4 q = wq[d4], v = wv[d4];
        const float g0 = s_g[b][d4 * 4 + 0], g1 = s_g[b][d4 * 4 + 1];
        const float g2 = s_g[b][d4 * 4 + 2], g3 = s_g[b][d4 * 4 + 3];
        aq += q.x * g0 + q.y * g1 + q.z * g2 + q.w * g3;
        av += v.x * g0 + v.y * g1 + v.z * g2 + v.w * g3;
    }
    qg[(size_t)b * Dn + o] = aq;
    vg[(size_t)b * Dn + o] = av;
}

// ---------------------------------------------------------------- k_Ah
// Ah[b,h,d] = sum_j qg[b,h*64+j] * Wk[h*64+j, d]  (f16; key bias dropped:
// softmax is shift-invariant). Grid (D/128, H); Wk read once across batches.
__global__ void k_Ah(const float* __restrict__ qg, const float* __restrict__ ipw,
                     half_t* __restrict__ Ah)
{
    __shared__ float s_q[Bn][HDn];
    __shared__ float s_part[256][9];
    const int dc = blockIdx.x, h = blockIdx.y;
    const int t = threadIdx.x;
    for (int r = 0; r < 2; ++r) {
        const int idx = r * 256 + t;                 // 0..511
        s_q[idx >> 6][idx & 63] = qg[(size_t)(idx >> 6) * Dn + h * HDn + (idx & 63)];
    }
    __syncthreads();

    const int d = dc * 128 + (t & 127);
    const int j0 = (t >> 7) * 32;
    float acc[Bn];
#pragma unroll
    for (int b = 0; b < Bn; ++b) acc[b] = 0.f;
    for (int j = 0; j < 32; ++j) {
        const float w = ipw[(size_t)(Dn + h * HDn + j0 + j) * Dn + d];
#pragma unroll
        for (int b = 0; b < Bn; ++b) acc[b] += s_q[b][j0 + j] * w;
    }
#pragma unroll
    for (int b = 0; b < Bn; ++b) s_part[t][b] = acc[b];
    __syncthreads();
    if (t < 128) {
#pragma unroll
        for (int b = 0; b < Bn; ++b) {
            const float tot = s_part[t][b] + s_part[t + 128][b];
            Ah[((size_t)(b * Hn + h) << 10) + dc * 128 + t] = (half_t)tot;
        }
    }
}

// ---------------------------------------------------------------- k_rowdot
// Batch-amortized GEMV: out[b,o] = W[o,:].x_b + bias[o], 1024-wide.
__global__ void k_rowdot(const float* __restrict__ W, const float* __restrict__ bias,
                         const float* __restrict__ x, float* __restrict__ out)
{
    __shared__ float s_x[Bn][VP];
    const int t = threadIdx.x;
    for (int r = 0; r < 32; ++r) {
        const int idx = r * 256 + t;
        s_x[idx >> 10][idx & 1023] = x[idx];
    }
    __syncthreads();

    const int o = blockIdx.x * 32 + (t >> 3);
    const int b = t & 7;
    const float4* wr = (const float4*)(W + (size_t)o * Dn);
    float acc = bias[o];
    for (int d4 = 0; d4 < Dn / 4; ++d4) {
        const float4 w = wr[d4];
        acc += w.x * s_x[b][d4 * 4 + 0] + w.y * s_x[b][d4 * 4 + 1] +
               w.z * s_x[b][d4 * 4 + 2] + w.w * s_x[b][d4 * 4 + 3];
    }
    out[(size_t)b * Dn + o] = acc;
}

// ---------------------------------------------------------------- k_scores
// scores[b,h,s] = seq[b,s,:].Ah[b,h,:] * 0.125 via WMMA f16, fragments direct
// from global. Block: 128 thr (4 waves), 64 s-rows. Grid (S/64, B) = 128 blocks.
__global__ void k_scores(const float* __restrict__ seq,
                         const half_t* __restrict__ Ah,
                         float* __restrict__ scores)
{
    const int b = blockIdx.y;
    const int s0 = blockIdx.x * 64;
    const int t = threadIdx.x;
    const int w = t >> 5;
    const int lane = t & 31;
    const int m = lane & 15;
    const int hs = lane >> 4;

    // A-operand row for this lane (seq), B-operand column row (Ah[h=m])
    const float*  arow = seq + ((size_t)(b * Sn + s0 + w * 16 + m) << 10);
    const half_t* brow = Ah + ((size_t)(b * Hn + m) << 10);

    v8f c = {};
    for (int d0 = 0; d0 < Dn; d0 += 32) {
        // A fragment: K = hs*16 + i (contiguous, matches B)
        const float4 a0 = *(const float4*)(arow + d0 + hs * 16);
        const float4 a1 = *(const float4*)(arow + d0 + hs * 16 + 4);
        const float4 a2 = *(const float4*)(arow + d0 + hs * 16 + 8);
        const float4 a3 = *(const float4*)(arow + d0 + hs * 16 + 12);
        v16h a;
        a[0] = (half_t)a0.x; a[1] = (half_t)a0.y; a[2]  = (half_t)a0.z; a[3]  = (half_t)a0.w;
        a[4] = (half_t)a1.x; a[5] = (half_t)a1.y; a[6]  = (half_t)a1.z; a[7]  = (half_t)a1.w;
        a[8] = (half_t)a2.x; a[9] = (half_t)a2.y; a[10] = (half_t)a2.z; a[11] = (half_t)a2.w;
        a[12] = (half_t)a3.x; a[13] = (half_t)a3.y; a[14] = (half_t)a3.z; a[15] = (half_t)a3.w;
        // B fragment: column n=m, K = hs*16 + i -> 16 contiguous halves (32B aligned)
        const v16h bm = *(const v16h*)(brow + d0 + hs * 16);
        c = __builtin_amdgcn_wmma_f32_16x16x32_f16(false, a, false, bm,
                                                   (short)0, c, false, false);
    }
    // C layout: lane -> N = lane&15 (=h), element g -> M = g + hs*8 (=s offset)
#pragma unroll
    for (int g = 0; g < 8; ++g) {
        const int srow = s0 + w * 16 + hs * 8 + g;
        scores[((size_t)(b * Hn + m) << 10) + srow] = c[g] * 0.125f;
    }
}

// ---------------------------------------------------------------- k_softmax
__global__ void k_softmax(const float* __restrict__ scores,
                          half_t* __restrict__ attn)
{
    __shared__ float red[256];
    const int b = blockIdx.y, h = blockIdx.x, t = threadIdx.x;
    const size_t base = (size_t)(b * Hn + h) << 10;
    float v[4];
    float lmax = -3.4e38f;
    for (int r = 0; r < 4; ++r) { v[r] = scores[base + r * 256 + t]; lmax = fmaxf(lmax, v[r]); }
    const float M = block_reduce_max(lmax, red);
    float e[4], ls = 0.f;
    for (int r = 0; r < 4; ++r) { e[r] = __expf(v[r] - M); ls += e[r]; }
    const float inv = 1.f / block_reduce_sum(ls, red);
    for (int r = 0; r < 4; ++r) attn[base + r * 256 + t] = (half_t)(e[r] * inv);
}

// ---------------------------------------------------------------- k_ctx
// ctx[b,h,d] = sum_s attn[b,h,s]*seq[b,s,d] via WMMA (M=h, N=d, K=s), fragments
// direct from global. Block: 128 thr (4 waves), 64 d-cols. Grid (D/64, B).
__global__ void k_ctx(const float* __restrict__ seq,
                      const half_t* __restrict__ attn,
                      float* __restrict__ ctx)
{
    const int b = blockIdx.y;
    const int dblk = blockIdx.x * 64;
    const int t = threadIdx.x;
    const int w = t >> 5;
    const int lane = t & 31;
    const int m = lane & 15;
    const int hs = lane >> 4;

    const half_t* arow = attn + ((size_t)(b * Hn + m) << 10);   // A: row h=m over K=s
    const int dcol = dblk + w * 16 + m;                          // B: column n=m

    v8f c = {};
    for (int s0 = 0; s0 < Sn; s0 += 32) {
        // A fragment: two contiguous 8-half runs (16B aligned)
        const v8h alo = *(const v8h*)(arow + s0 + hs * 8);
        const v8h ahi = *(const v8h*)(arow + s0 + 16 + hs * 8);
        v16h a;
#pragma unroll
        for (int i = 0; i < 8; ++i) { a[i] = alo[i]; a[i + 8] = ahi[i]; }
        // B fragment: K = hs*8 + i (i<8), 16 + hs*8 + (i-8) (i>=8) rows of seq at column dcol (f32 gather)
        v16h bm;
#pragma unroll
        for (int i = 0; i < 16; ++i) {
            const int srow = s0 + hs * 8 + i + (i >= 8 ? 8 : 0);
            bm[i] = (half_t)seq[((size_t)(b * Sn + srow) << 10) + dcol];
        }
        c = __builtin_amdgcn_wmma_f32_16x16x32_f16(false, a, false, bm,
                                                   (short)0, c, false, false);
    }
#pragma unroll
    for (int g = 0; g < 8; ++g) {
        const int h = g + hs * 8;
        ctx[((size_t)(b * Hn + h) << 10) + dcol] = c[g];
    }
}

// ---------------------------------------------------------------- k_vctx
// vctx[b,o] = Wv[o,:].ctx[b, o/64, :] + bv[o]. Block rows share one head h.
__global__ void k_vctx(const float* __restrict__ ctx,
                       const float* __restrict__ ipw, const float* __restrict__ ipb,
                       float* __restrict__ vctx)
{
    __shared__ float s_c[Bn][VP];
    const int t = threadIdx.x;
    const int o0 = blockIdx.x * 32;
    const int h = o0 >> 6;                           // constant within block
    for (int r = 0; r < 32; ++r) {
        const int idx = r * 256 + t;                 // 0..8191
        s_c[idx >> 10][idx & 1023] = ctx[((size_t)((idx >> 10) * Hn + h) << 10) + (idx & 1023)];
    }
    __syncthreads();

    const int o = o0 + (t >> 3);
    const int b = t & 7;
    const float4* wr = (const float4*)(ipw + (size_t)(2 * Dn + o) * Dn);
    float acc = ipb[2 * Dn + o];
    for (int d4 = 0; d4 < Dn / 4; ++d4) {
        const float4 w = wr[d4];
        acc += w.x * s_c[b][d4 * 4 + 0] + w.y * s_c[b][d4 * 4 + 1] +
               w.z * s_c[b][d4 * 4 + 2] + w.w * s_c[b][d4 * 4 + 3];
    }
    vctx[(size_t)b * Dn + o] = acc;
}

// ---------------------------------------------------------------- k_fuse
// gate/proj rows (2048-wide) vs combined=[satt||gatt]; x = proj + gate*satt + (1-gate)*gatt
__global__ void k_fuse(const float* __restrict__ satt, const float* __restrict__ gatt,
                       const float* __restrict__ gatew, const float* __restrict__ gateb,
                       const float* __restrict__ projw, const float* __restrict__ projb,
                       float* __restrict__ xbuf)
{
    __shared__ float s_s[Bn][VP];
    __shared__ float s_t[Bn][VP];
    const int t = threadIdx.x;
    for (int r = 0; r < 32; ++r) {
        const int idx = r * 256 + t;
        s_s[idx >> 10][idx & 1023] = satt[idx];
        s_t[idx >> 10][idx & 1023] = gatt[idx];
    }
    __syncthreads();

    const int o = blockIdx.x * 32 + (t >> 3);
    const int b = t & 7;
    const float4* gw = (const float4*)(gatew + (size_t)o * 2 * Dn);
    const float4* pw = (const float4*)(projw + (size_t)o * 2 * Dn);
    float ga = gateb[o], pa = projb[o];
    for (int d4 = 0; d4 < Dn / 4; ++d4) {
        const float4 g = gw[d4], p = pw[d4];
        const float c0 = s_s[b][d4 * 4 + 0], c1 = s_s[b][d4 * 4 + 1];
        const float c2 = s_s[b][d4 * 4 + 2], c3 = s_s[b][d4 * 4 + 3];
        ga += g.x * c0 + g.y * c1 + g.z * c2 + g.w * c3;
        pa += p.x * c0 + p.y * c1 + p.z * c2 + p.w * c3;
    }
    for (int d4 = 0; d4 < Dn / 4; ++d4) {
        const float4 g = gw[Dn / 4 + d4], p = pw[Dn / 4 + d4];
        const float c0 = s_t[b][d4 * 4 + 0], c1 = s_t[b][d4 * 4 + 1];
        const float c2 = s_t[b][d4 * 4 + 2], c3 = s_t[b][d4 * 4 + 3];
        ga += g.x * c0 + g.y * c1 + g.z * c2 + g.w * c3;
        pa += p.x * c0 + p.y * c1 + p.z * c2 + p.w * c3;
    }
    const float gate = 1.f / (1.f + __expf(-ga));
    xbuf[(size_t)b * Dn + o] = pa + gate * s_s[b][o] + (1.f - gate) * s_t[b][o];
}

// ---------------------------------------------------------------- k_ln
__global__ void k_ln(const float* __restrict__ xbuf,
                     const float* __restrict__ lng, const float* __restrict__ lnb,
                     float* __restrict__ y)
{
    __shared__ float red[256];
    const int b = blockIdx.x, t = threadIdx.x;
    float xv[4], ls = 0.f, lq = 0.f;
    for (int r = 0; r < 4; ++r) {
        xv[r] = xbuf[(size_t)b * Dn + r * 256 + t];
        ls += xv[r]; lq += xv[r] * xv[r];
    }
    const float sum = block_reduce_sum(ls, red);
    const float sq  = block_reduce_sum(lq, red);
    const float mu = sum * (1.f / Dn);
    const float var = sq * (1.f / Dn) - mu * mu;
    const float inv = rsqrtf(var + EPS);
    for (int r = 0; r < 4; ++r) {
        const int o = r * 256 + t;
        y[(size_t)b * Dn + o] = (xv[r] - mu) * inv * lng[o] + lnb[o];
    }
}

// ---------------------------------------------------------------- k_bcast
// 32MB write-once stream: non-temporal stores keep L2 for seq/weights.
__global__ void k_bcast(const float* __restrict__ y, float* __restrict__ out)
{
    const size_t lin4 = (size_t)blockIdx.x * blockDim.x + threadIdx.x;
    if (lin4 >= (size_t)Bn * Sn * Dn / 4) return;
    const int d4 = (int)(lin4 & 255);
    const int b = (int)(lin4 >> 18);
    const v4f v = ((const v4f*)y)[b * (Dn / 4) + d4];
    __builtin_nontemporal_store(v, (v4f*)out + lin4);
}

// ---------------------------------------------------------------- launch
extern "C" void kernel_launch(void* const* d_in, const int* in_sizes, int n_in,
                              void* d_out, int out_size, void* d_ws, size_t ws_size,
                              hipStream_t stream) {
    const float* seq    = (const float*)d_in[0];
    const float* graph  = (const float*)d_in[1];
    const float* ipw    = (const float*)d_in[2];
    const float* ipb    = (const float*)d_in[3];
    const float* outw   = (const float*)d_in[4];
    const float* outb   = (const float*)d_in[5];
    const float* gatew  = (const float*)d_in[6];
    const float* gateb  = (const float*)d_in[7];
    const float* projw  = (const float*)d_in[8];
    const float* projb  = (const float*)d_in[9];
    const float* lng    = (const float*)d_in[10];
    const float* lnb    = (const float*)d_in[11];
    float* out = (float*)d_out;
    (void)in_sizes; (void)n_in; (void)out_size; (void)ws_size;

    char* w = (char*)d_ws;
    size_t off = 0;
    auto alloc_f = [&](size_t n) { float*  p = (float*)(w + off);  off += n * 4; return p; };
    auto alloc_h = [&](size_t n) { half_t* p = (half_t*)(w + off); off += n * 2; return p; };
    float*  qg     = alloc_f((size_t)Bn * Dn);
    float*  vg     = alloc_f((size_t)Bn * Dn);
    float*  satt   = alloc_f((size_t)Bn * Dn);
    half_t* Ah     = alloc_h((size_t)Bn * Hn * Dn);
    float*  scores = alloc_f((size_t)Bn * Hn * Sn);
    half_t* attn   = alloc_h((size_t)Bn * Hn * Sn);
    float*  ctx    = alloc_f((size_t)Bn * Hn * Dn);
    float*  vctx   = alloc_f((size_t)Bn * Dn);
    float*  gatt   = alloc_f((size_t)Bn * Dn);
    float*  xbuf   = alloc_f((size_t)Bn * Dn);
    float*  y      = alloc_f((size_t)Bn * Dn);

    k_qgvg<<<32, 256, 0, stream>>>(graph, ipw, ipb, qg, vg);
    k_Ah<<<dim3(Dn / 128, Hn), 256, 0, stream>>>(qg, ipw, Ah);
    k_rowdot<<<32, 256, 0, stream>>>(outw, outb, vg, satt);          // seq_att vector
    k_scores<<<dim3(Sn / 64, Bn), 128, 0, stream>>>(seq, Ah, scores);
    k_softmax<<<dim3(Hn, Bn), 256, 0, stream>>>(scores, attn);
    k_ctx<<<dim3(Dn / 64, Bn), 128, 0, stream>>>(seq, attn, ctx);
    k_vctx<<<32, 256, 0, stream>>>(ctx, ipw, ipb, vctx);
    k_rowdot<<<32, 256, 0, stream>>>(outw, outb, vctx, gatt);        // graph_att vector
    k_fuse<<<32, 256, 0, stream>>>(satt, gatt, gatew, gateb, projw, projb, xbuf);
    k_ln<<<Bn, 256, 0, stream>>>(xbuf, lng, lnb, y);
    k_bcast<<<(Bn * Sn * Dn / 4 + 255) / 256, 256, 0, stream>>>(y, out);
}
